// BipartiteGraphMultiHeadAttentionLayer_12541304504946
// MI455X (gfx1250) — matrix operand order repaired and checked by reading, not compiled, not run on hardware
//
#include <hip/hip_runtime.h>
#include <hip/hip_bf16.h>

// ---------------------------------------------------------------------------
// Bipartite multi-head GAT for MI455X (gfx1250, wave32, WMMA bf16 path)
// B=2, N_F=2048, N_E=4096, FDIM=EDIM=128, HDIM=64, HEADS=4
// ---------------------------------------------------------------------------

typedef __attribute__((ext_vector_type(16))) __bf16 v16bf;
typedef __attribute__((ext_vector_type(8)))  float  v8f;

#define KB 2
#define KNF 2048
#define KNE 4096
#define KFD 128
#define KHD 64
#define KH  4
#define KNEG_INF (-1.0e9f)

// A-fragment K pattern for 16-bit 16x16x32 WMMA (ISA 7.12.2):
// lanes 0-15: V0..3 -> K=0,2,4,6 pairs ; V4..7 -> K=16,18,20,22 pairs
// lanes 16-31: +8
__device__ __forceinline__ int kpatA(int v, int h16) {
    return ((v & 3) * 2) + ((v >> 2) * 16) + (h16 * 8);
}

__device__ __forceinline__ float leaky02(float s) {
    return (s > 0.f) ? s : 0.2f * s;
}

// ---------------------------------------------------------------------------
// Transpose W [H, K, 64] -> WT [H, 64, K]   (K = 128)
// ---------------------------------------------------------------------------
__global__ void wtrans_kernel(const float* __restrict__ W,
                              float* __restrict__ WT) {
    int idx = blockIdx.x * blockDim.x + threadIdx.x;
    int total = KH * KFD * KHD;
    if (idx >= total) return;
    int h   = idx / (KFD * KHD);
    int rem = idx - h * (KFD * KHD);
    int f   = rem / KHD;
    int d   = rem - f * KHD;
    WT[((size_t)h * KHD + d) * KFD + f] = W[idx];
}

// ---------------------------------------------------------------------------
// Projection: H1[b,h,n,d] = X[b,n,:] @ W[h,:,d]  via v_wmma_f32_16x16x32_bf16
// One wave handles a 16(node) x 64(d) output tile; 4 K-chunks of 32.
// Also emits bf16 d-major transposed copy H1bT[b,h,d,n] for the aggregation B
// operand (consecutive-K pairs contiguous in memory).
// ---------------------------------------------------------------------------
__global__ __launch_bounds__(128)
void proj_kernel(const float* __restrict__ X,    // [B, N, 128]
                 const float* __restrict__ WT,   // [H, 64, 128]
                 float* __restrict__ H1,         // [B,H,N,64]
                 __bf16* __restrict__ H1bT,      // [B,H,64,N]
                 int N) {
    int wave = (blockIdx.x << 2) + (threadIdx.x >> 5);
    int lane = threadIdx.x & 31;
    int tilesPerBH = N >> 4;
    int bh   = wave / tilesPerBH;
    int tile = wave - bh * tilesPerBH;
    int b = bh >> 2;
    int h = bh & 3;
    int nodeBase = tile << 4;
    int r   = lane & 15;
    int h16 = lane >> 4;

    const float* Xb = X  + (size_t)b * N * KFD;
    const float* Wb = WT + (size_t)h * KHD * KFD;

    v8f acc[4];
    {
        v8f z = {};
        acc[0] = z; acc[1] = z; acc[2] = z; acc[3] = z;
    }

    for (int kc = 0; kc < KFD; kc += 32) {
        v16bf afrag;
#pragma unroll
        for (int v = 0; v < 8; ++v) {
            int k0 = kc + kpatA(v, h16);
            float2 x2 = *(const float2*)(Xb + (size_t)(nodeBase + r) * KFD + k0);
            afrag[2 * v]     = (__bf16)x2.x;
            afrag[2 * v + 1] = (__bf16)x2.y;
        }
#pragma unroll
        for (int nt = 0; nt < 4; ++nt) {
            v16bf bfrag;
            int n = nt * 16 + r;
#pragma unroll
            for (int v = 0; v < 8; ++v) {
                int k0 = kc + kpatA(v, h16);
                float2 w2 = *(const float2*)(Wb + (size_t)n * KFD + k0);
                bfrag[2 * v]     = (__bf16)w2.x;
                bfrag[2 * v + 1] = (__bf16)w2.y;
            }
            acc[nt] = __builtin_amdgcn_wmma_f32_16x16x32_bf16(
                false, afrag, false, bfrag, (short)0, acc[nt], false, false);
        }
    }

    float*  H1b = H1   + (size_t)bh * N * KHD;
    __bf16* HTb = H1bT + (size_t)bh * KHD * N;
#pragma unroll
    for (int nt = 0; nt < 4; ++nt) {
        int d = nt * 16 + r;
#pragma unroll
        for (int v = 0; v < 8; ++v) {
            int node = nodeBase + v + 8 * h16;
            float val = acc[nt][v];
            H1b[(size_t)node * KHD + d] = val;
            HTb[(size_t)d * N + node]   = (__bf16)val;
        }
    }
}

// ---------------------------------------------------------------------------
// Attention score vectors: e0 = H1 . a0[h], e1 = H1 . a1[h]  (float4 loads)
// ---------------------------------------------------------------------------
__global__ void dot_kernel(const float* __restrict__ H1, int N,
                           const float* __restrict__ a0,
                           const float* __restrict__ a1,
                           float* __restrict__ e0,
                           float* __restrict__ e1) {
    int idx = blockIdx.x * blockDim.x + threadIdx.x;  // over B*H*N
    if (idx >= KB * KH * N) return;
    int bh = idx / N;
    int h  = bh & 3;
    const float4* row = (const float4*)(H1 + (size_t)idx * KHD);
    const float4* p0  = (const float4*)(a0 + h * KHD);
    const float4* p1  = (const float4*)(a1 + h * KHD);
    float s0 = 0.f, s1 = 0.f;
#pragma unroll
    for (int q = 0; q < KHD / 4; ++q) {
        float4 x = row[q];
        float4 u = p0[q];
        float4 w = p1[q];
        s0 += x.x * u.x + x.y * u.y + x.z * u.z + x.w * u.w;
        s1 += x.x * w.x + x.y * w.y + x.z * w.z + x.w * w.w;
    }
    e0[idx] = s0;
    e1[idx] = s1;
}

// ---------------------------------------------------------------------------
// Fused masked-softmax attention + aggregation (flash style, two pass).
// One wave per 16 source rows; lane pair (L, L+16) shares row r = L&15.
// Pass 1: blockwise online row max / sum over all dst (int4/float4 loads,
//         each lane-half owns contiguous 8-blocks, combined via shfl_xor 16).
// Pass 2: p = exp(s - max) packed to bf16 A-frags; 4 WMMAs per 32-dst tile
//         against d-major bf16 h_dst.
// Epilogue: divide by row sum (shfl broadcast), optional relu, scatter into
// the [B, Nsrc, HEADS*64] concat layout.
// ---------------------------------------------------------------------------
__global__ __launch_bounds__(128)
void attn_kernel(const float* __restrict__ eSrc,   // [B,H,Nsrc]
                 const float* __restrict__ eDst,   // [B,H,Ndst]
                 const int*   __restrict__ adj,    // [Nsrc, Ndst]
                 const __bf16* __restrict__ HdT,   // [B,H,64,Ndst] bf16
                 float* __restrict__ outBase,      // [B, Nsrc, 256]
                 int Nsrc, int Ndst, int applyRelu) {
    int wave = (blockIdx.x << 2) + (threadIdx.x >> 5);
    int lane = threadIdx.x & 31;
    int tilesPerBH = Nsrc >> 4;
    int bh   = wave / tilesPerBH;
    int tile = wave - bh * tilesPerBH;
    int b = bh >> 2;
    int h = bh & 3;
    int srcBase = tile << 4;
    int r   = lane & 15;
    int h16 = lane >> 4;
    int src = srcBase + r;

    const float* ed     = eDst + (size_t)bh * Ndst;
    const int*   adjRow = adj  + (size_t)src * Ndst;
    float es = eSrc[(size_t)bh * Nsrc + src];

    // ---- pass 1: online row max / sum; lane-half owns 8-blocks (16B loads) ----
    float mrun = -3.0e38f;
    float lrun = 0.0f;
    for (int mb = 0; mb < Ndst; mb += 16) {
        int m0 = mb + h16 * 8;                     // 8-aligned -> int4 aligned
        int4   a0 = *(const int4*)(adjRow + m0);
        int4   a1 = *(const int4*)(adjRow + m0 + 4);
        float4 f0 = *(const float4*)(ed + m0);
        float4 f1 = *(const float4*)(ed + m0 + 4);
        float s[8];
        s[0] = (a0.x == 0) ? KNEG_INF : leaky02(es + f0.x);
        s[1] = (a0.y == 0) ? KNEG_INF : leaky02(es + f0.y);
        s[2] = (a0.z == 0) ? KNEG_INF : leaky02(es + f0.z);
        s[3] = (a0.w == 0) ? KNEG_INF : leaky02(es + f0.w);
        s[4] = (a1.x == 0) ? KNEG_INF : leaky02(es + f1.x);
        s[5] = (a1.y == 0) ? KNEG_INF : leaky02(es + f1.y);
        s[6] = (a1.z == 0) ? KNEG_INF : leaky02(es + f1.z);
        s[7] = (a1.w == 0) ? KNEG_INF : leaky02(es + f1.w);
        float bm = s[0];
#pragma unroll
        for (int j = 1; j < 8; ++j) bm = fmaxf(bm, s[j]);
        float nm = fmaxf(mrun, bm);
        float part = 0.f;
#pragma unroll
        for (int j = 0; j < 8; ++j) part += __expf(s[j] - nm);
        lrun = lrun * __expf(mrun - nm) + part;
        mrun = nm;
    }
    float mo = __shfl_xor(mrun, 16, 32);
    float lo = __shfl_xor(lrun, 16, 32);
    float M    = fmaxf(mrun, mo);
    float Lsum = lrun * __expf(mrun - M) + lo * __expf(mo - M);
    float rinv = 1.0f / Lsum;

    // ---- pass 2: WMMA aggregation of p @ h_dst ----
    v8f acc[4];
    {
        v8f z = {};
        acc[0] = z; acc[1] = z; acc[2] = z; acc[3] = z;
    }
    const __bf16* Hd = HdT + (size_t)bh * KHD * Ndst;

    for (int dstBase = 0; dstBase < Ndst; dstBase += 32) {
        if (dstBase + 32 < Ndst) {
            // prefetch next tile's score inputs (global_prefetch_b8)
            __builtin_prefetch(adjRow + dstBase + 32 + h16 * 16, 0, 0);
            __builtin_prefetch(ed + dstBase + 32 + h16 * 16, 0, 0);
        }
        v16bf afrag;
#pragma unroll
        for (int v = 0; v < 8; ++v) {
            int k0 = dstBase + kpatA(v, h16);
            int2   a2 = *(const int2*)(adjRow + k0);
            float2 e2 = *(const float2*)(ed + k0);
            float s0 = leaky02(es + e2.x);
            float s1 = leaky02(es + e2.y);
            if (a2.x == 0) s0 = KNEG_INF;
            if (a2.y == 0) s1 = KNEG_INF;
            afrag[2 * v]     = (__bf16)__expf(s0 - M);
            afrag[2 * v + 1] = (__bf16)__expf(s1 - M);
        }
#pragma unroll
        for (int nt = 0; nt < 4; ++nt) {
            v16bf bfrag;
            const __bf16* col =
                Hd + (size_t)(nt * 16 + r) * Ndst + dstBase;
#pragma unroll
            for (int v = 0; v < 8; ++v) {
                union { unsigned int u; __bf16 hh[2]; } c;
                c.u = *(const unsigned int*)(col + kpatA(v, h16));  // 4B aligned
                bfrag[2 * v]     = c.hh[0];
                bfrag[2 * v + 1] = c.hh[1];
            }
            acc[nt] = __builtin_amdgcn_wmma_f32_16x16x32_bf16(
                false, afrag, false, bfrag, (short)0, acc[nt], false, false);
        }
    }

    // ---- epilogue: normalize, optional relu, scatter to concat layout ----
#pragma unroll
    for (int v = 0; v < 8; ++v) {
        int m = v + 8 * h16;
        float sc = __shfl(rinv, m, 32);   // row-m stats live in lane m
        int node = srcBase + m;
#pragma unroll
        for (int nt = 0; nt < 4; ++nt) {
            int d = nt * 16 + r;
            float val = acc[nt][v] * sc;
            if (applyRelu) val = fmaxf(val, 0.f);
            outBase[((size_t)b * Nsrc + node) * (KH * KHD) + h * KHD + d] = val;
        }
    }
}

// ---------------------------------------------------------------------------
// Row softmax over 256 features (out_fe epilogue)
// ---------------------------------------------------------------------------
__global__ void softmax256_kernel(const float* __restrict__ in,
                                  float* __restrict__ out) {
    __shared__ float red[256];
    size_t row = blockIdx.x;
    int t = threadIdx.x;
    float v = in[row * 256 + t];
    red[t] = v;
    __syncthreads();
    for (int s = 128; s > 0; s >>= 1) {
        if (t < s) red[t] = fmaxf(red[t], red[t + s]);
        __syncthreads();
    }
    float mx = red[0];
    __syncthreads();
    float e = __expf(v - mx);
    red[t] = e;
    __syncthreads();
    for (int s = 128; s > 0; s >>= 1) {
        if (t < s) red[t] += red[t + s];
        __syncthreads();
    }
    out[row * 256 + t] = e / red[0];
}

// ---------------------------------------------------------------------------
// Launcher
// ---------------------------------------------------------------------------
extern "C" void kernel_launch(void* const* d_in, const int* in_sizes, int n_in,
                              void* d_out, int out_size, void* d_ws, size_t ws_size,
                              hipStream_t stream) {
    (void)in_sizes; (void)n_in; (void)out_size; (void)ws_size;

    const float* F0     = (const float*)d_in[0];
    const float* E0     = (const float*)d_in[1];
    const int*   adj_fe = (const int*)d_in[2];
    const int*   adj_ef = (const int*)d_in[3];
    const float* Wf     = (const float*)d_in[4];
    const float* We     = (const float*)d_in[5];
    const float* a_fe   = (const float*)d_in[6];
    const float* a_ef   = (const float*)d_in[7];

    // ---- workspace carve-up (bytes) ----
    char* ws = (char*)d_ws;
    float*  WfT     = (float*) (ws + 0);          //   4*64*128 f32 = 128 KB
    float*  WeT     = (float*) (ws + 131072);     //   128 KB
    float*  F1      = (float*) (ws + 262144);     //   2*4*2048*64 f32 = 4 MB
    float*  E1      = (float*) (ws + 4456448);    //   8 MB
    __bf16* F1bT    = (__bf16*)(ws + 12845056);   //   2 MB (bf16, d-major)
    __bf16* E1bT    = (__bf16*)(ws + 14942208);   //   4 MB
    float*  e_srcfe = (float*) (ws + 19136512);   //   [B,H,N_F]
    float*  e_dstfe = (float*) (ws + 19202048);   //   [B,H,N_E]
    float*  e_srcef = (float*) (ws + 19333120);   //   [B,H,N_E]
    float*  e_dstef = (float*) (ws + 19464192);   //   [B,H,N_F]
    float*  hp_fe   = (float*) (ws + 19529728);   //   [B,N_F,256] = 4 MB

    float* out_fe = (float*)d_out;                          // [B,N_F,256]
    float* out_ef = (float*)d_out + (size_t)KB * KNF * 256; // [B,N_E,256]

    // 1) transpose W kernels for contiguous B-fragment loads
    wtrans_kernel<<<(KH * KFD * KHD + 255) / 256, 256, 0, stream>>>(Wf, WfT);
    wtrans_kernel<<<(KH * KFD * KHD + 255) / 256, 256, 0, stream>>>(We, WeT);

    // 2) projections (WMMA bf16): waves = B*H*(N/16); 4 waves per block
    proj_kernel<<<(KB * KH * (KNF / 16)) / 4, 128, 0, stream>>>(F0, WfT, F1, F1bT, KNF);
    proj_kernel<<<(KB * KH * (KNE / 16)) / 4, 128, 0, stream>>>(E0, WeT, E1, E1bT, KNE);

    // 3) attention logit vectors
    dot_kernel<<<(KB * KH * KNF + 255) / 256, 256, 0, stream>>>(
        F1, KNF, a_fe, a_ef, e_srcfe, e_dstef);
    dot_kernel<<<(KB * KH * KNE + 255) / 256, 256, 0, stream>>>(
        E1, KNE, a_fe, a_ef, e_dstfe, e_srcef);

    // 4) fused masked softmax + WMMA aggregation
    //    F->E direction: raw output to workspace (softmax epilogue follows)
    attn_kernel<<<(KB * KH * (KNF / 16)) / 4, 128, 0, stream>>>(
        e_srcfe, e_dstfe, adj_fe, E1bT, hp_fe, KNF, KNE, /*relu=*/0);
    //    E->F direction: relu fused, straight into d_out
    attn_kernel<<<(KB * KH * (KNE / 16)) / 4, 128, 0, stream>>>(
        e_srcef, e_dstef, adj_ef, F1bT, out_ef, KNE, KNF, /*relu=*/1);

    // 5) out_fe = softmax over the 256 concat features
    softmax256_kernel<<<KB * KNF, 256, 0, stream>>>(hp_fe, out_fe);
}
